// DecoderRNN_54236847014058
// MI455X (gfx1250) — compile-verified, matrix operand order checked by reading, not run on hardware
//
#include <hip/hip_runtime.h>
#include <stdint.h>

typedef __attribute__((ext_vector_type(16))) _Float16 v16h;
typedef __attribute__((ext_vector_type(8)))  float    v8f;
typedef __attribute__((ext_vector_type(4)))  unsigned u32x4;
typedef __attribute__((ext_vector_type(8)))  int      i32x8;
typedef __attribute__((ext_vector_type(4)))  int      i32x4;

#define HD     512
#define G3     1536
#define BATCH  64
#define TSTEPS 1024
#define KT     16          // 512 / 32 k-tiles
#define NB     64          // persistent blocks
#define TPB    256         // 8 wave32 per block
#define WPB    (TPB / 32)
#define NWAVE  (NB * WPB)

struct Params {
  const float *ctx, *Wih0, *Whh0, *bih0, *bhh0, *Wih1, *Whh1, *bih1, *bhh1;
  const float *Wo1, *bo1, *Wo2, *bo2;
  unsigned *flags;
  float *prev, *h0, *h1, *y1, *gh0, *gh1, *gi1, *gi0c, *wc;
  _Float16 *h0f, *h1f, *ctxf, *Wih0f, *Whh0f, *Wih1f, *Whh1f, *Wo1f;
  float *out;
};

// WMMA 16x16x32 f16 fragment layout (CDNA5 ISA 7.12.2):
// lane L holds row (A) / col (B) = L%16; its 16 halves cover
//   e=0..7  -> K = (L/16)*8 + e ; e=8..15 -> K = 16 + (L/16)*8 + (e-8)
// Each (tile_row, tile_k) fragment stored as 32 lanes x 16 halves contiguous,
// so a lane's slice is one aligned 32-byte load.
__device__ __forceinline__ int fragIndex(int row, int k, int ktiles) {
  int rt = row >> 4;
  int kt = k >> 5;
  int kr = k & 31;
  int L  = (((kr & 15) >> 3) << 4) | (row & 15);
  int e  = ((kr >> 4) << 3) | (k & 7);
  return ((((rt * ktiles) + kt) * 32 + L) << 4) + e;
}

__device__ __forceinline__ v8f wmma_acc(v16h a, v16h b, v8f c) {
  return __builtin_amdgcn_wmma_f32_16x16x32_f16(false, a, false, b,
                                                (short)0, c, false, false);
}

__device__ __forceinline__ float sigmoidf_(float x) {
  return 1.0f / (1.0f + __expf(-x));
}

// ---- Tensor Data Mover: DMA one contiguous run (ndwords x 4B) global->LDS ----
// D# group0: count=1 | lds_addr | global_addr[56:0] | type=2 ("image")
// D# group1: data_size=4B, 1 row tensor: dim0 = tile_dim0 = stride0 = ndwords
// 6-arg builtin form (clang-23 / therock headers): groups 0..3 + extra + cpol.
__device__ __forceinline__ void tdm_load_lds_1d(void* lds_dst, const void* gsrc,
                                                int ndwords) {
  unsigned lds = (unsigned)(uintptr_t)lds_dst;          // low 32b = LDS offset
  unsigned long long ga = (unsigned long long)(uintptr_t)gsrc;
  u32x4 g0;
  g0[0] = 1u;                                           // count = 1 (valid D#)
  g0[1] = lds;                                          // lds_addr
  g0[2] = (unsigned)ga;                                 // global_addr[31:0]
  g0[3] = (unsigned)((ga >> 32) & 0x01FFFFFFull) | (2u << 30); // ga[56:32]|type=2
  i32x8 g1;
  g1[0] = 2 << 16;                                      // data_size = 4 bytes
  g1[1] = (ndwords & 0xffff) << 16;                     // tensor_dim0[15:0]
  g1[2] = ((ndwords >> 16) & 0xffff) | (1 << 16);       // dim0[31:16] | dim1 = 1
  g1[3] = (ndwords & 0xffff) << 16;                     // tile_dim0 = ndwords
  g1[4] = 1;                                            // tile_dim1=1, tile_dim2=0
  g1[5] = ndwords;                                      // tensor_dim0_stride lo32
  g1[6] = 0;                                            // stride0 hi | stride1 lo
  g1[7] = 0;
  i32x4 z4 = {0, 0, 0, 0};                              // groups 2/3 unused (<=2D)
  i32x8 z8 = {0, 0, 0, 0, 0, 0, 0, 0};
  __builtin_amdgcn_tensor_load_to_lds(g0, g1, z4, z4, z8, 0);
}

// Device-wide barrier for the persistent kernel (flags zeroed per launch).
__device__ __forceinline__ void gsync(unsigned* flags, unsigned e) {
  __syncthreads();
  __threadfence();
  if (threadIdx.x == 0)
    __hip_atomic_store(&flags[blockIdx.x], e, __ATOMIC_RELEASE,
                       __HIP_MEMORY_SCOPE_AGENT);
  if (threadIdx.x < NB) {
    while (__hip_atomic_load(&flags[threadIdx.x], __ATOMIC_ACQUIRE,
                             __HIP_MEMORY_SCOPE_AGENT) < e)
      __builtin_amdgcn_s_sleep(1);
  }
  __syncthreads();
}

// ---------------- one-time prep: f32 -> f16 fragment swizzle -----------------
__global__ void k_swizzle(Params p) {
  const int stride = gridDim.x * blockDim.x;
  const int t0 = blockIdx.x * blockDim.x + threadIdx.x;
  for (int i = t0; i < G3 * HD; i += stride) {
    int n = i >> 9, k = i & 511;
    int d = fragIndex(n, k, KT);
    p.Wih0f[d] = (_Float16)p.Wih0[n * (HD + 1) + k];   // row stride 513
    p.Whh0f[d] = (_Float16)p.Whh0[i];
    p.Wih1f[d] = (_Float16)p.Wih1[i];
    p.Whh1f[d] = (_Float16)p.Whh1[i];
  }
  for (int i = t0; i < HD * HD; i += stride) {
    int n = i >> 9, k = i & 511;
    p.Wo1f[fragIndex(n, k, KT)] = (_Float16)p.Wo1[i];
  }
  for (int i = t0; i < BATCH * HD; i += stride) {
    int b = i >> 9, k = i & 511;
    p.ctxf[fragIndex(b, k, KT)] = (_Float16)p.ctx[i];
  }
  for (int i = t0; i < G3; i += stride)
    p.wc[i] = p.Wih0[i * (HD + 1) + HD];               // last column (prev coeff)
}

// ------------- one-time: gi0c = ctx @ Wih0[:, :512]^T + b_ih0 ---------------
__global__ void k_gemm_ctx(Params p) {
  const int lane = threadIdx.x & 31;
  const int gw = blockIdx.x * (blockDim.x >> 5) + (threadIdx.x >> 5);
  const int nw = gridDim.x * (blockDim.x >> 5);
  for (int u = gw; u < 4 * 48; u += nw) {              // 2-wide N blocking
    int mt = u / 48, nt0 = (u % 48) * 2;
    v8f acc0 = {}, acc1 = {};
#pragma unroll
    for (int kt = 0; kt < KT; ++kt) {
      v16h a  = *(const v16h*)(p.ctxf  + ((((mt * KT) + kt) * 32 + lane) << 4));
      v16h b0 = *(const v16h*)(p.Wih0f + (((((nt0    ) * KT) + kt) * 32 + lane) << 4));
      v16h b1 = *(const v16h*)(p.Wih0f + (((((nt0 + 1) * KT) + kt) * 32 + lane) << 4));
      acc0 = wmma_acc(a, b0, acc0);
      acc1 = wmma_acc(a, b1, acc1);
    }
    int lg = lane >> 4, ln = lane & 15;
#pragma unroll
    for (int r = 0; r < 8; ++r) {
      int m = mt * 16 + r + 8 * lg;
      int n0 = nt0 * 16 + ln;
      p.gi0c[m * G3 + n0]      = acc0[r] + p.bih0[n0];
      p.gi0c[m * G3 + n0 + 16] = acc1[r] + p.bih0[n0 + 16];
    }
  }
}

// ------------------- persistent sequence kernel (1024 steps) -----------------
// LDS layout: sh[0..32767]      = h0 activation panel (f16 frags, 64 KB)
//             sh[32768..65535]  = h1 activation panel (f16 frags, 64 KB)
// Panels staged by TDM at phase 3 (h0n) / phase 5 (h1n) remain valid for the
// NEXT step's phase 1, so phase 1 needs no staging after the initial one.
__global__ void __launch_bounds__(TPB, 1) k_main(Params p) {
  extern __shared__ _Float16 sh[];
  const int tid  = threadIdx.x;
  const int lane = tid & 31;
  const int gw   = blockIdx.x * WPB + (tid >> 5);
  const int gtid = blockIdx.x * TPB + tid;
  unsigned ep = 0;

  // one-time stage of the (zeroed) initial h0/h1 panels
  if (tid < 32) {
    tdm_load_lds_1d(sh,         p.h0f, 16384);
    tdm_load_lds_1d(sh + 32768, p.h1f, 16384);
    __builtin_amdgcn_s_wait_tensorcnt(0);
  }
  __syncthreads();

  for (int t = 0; t < TSTEPS; ++t) {
    // ---- phase 1: gh0 = h0 @ Whh0^T ; gh1 = h1 @ Whh1^T (panels resident) ----
    for (int u = gw; u < 2 * 4 * 48; u += NWAVE) {
      int which = u >= 192;
      int idx = which ? u - 192 : u;
      int mt = idx / 48, nt0 = (idx % 48) * 2;
      const _Float16* A  = sh + (which ? 32768 : 0);
      const _Float16* Bf = which ? p.Whh1f : p.Whh0f;
      float* O = which ? p.gh1 : p.gh0;
      v8f acc0 = {}, acc1 = {};
#pragma unroll
      for (int kt = 0; kt < KT; ++kt) {
        v16h a  = *(const v16h*)(A  + ((((mt * KT) + kt) * 32 + lane) << 4));
        v16h b0 = *(const v16h*)(Bf + (((((nt0    ) * KT) + kt) * 32 + lane) << 4));
        v16h b1 = *(const v16h*)(Bf + (((((nt0 + 1) * KT) + kt) * 32 + lane) << 4));
        if (kt + 2 < KT)
          __builtin_prefetch((const void*)(Bf + (((((nt0) * KT) + kt + 2) * 32 + lane) << 4)), 0, 1);
        acc0 = wmma_acc(a, b0, acc0);
        acc1 = wmma_acc(a, b1, acc1);
      }
      int lg = lane >> 4, ln = lane & 15;
#pragma unroll
      for (int r = 0; r < 8; ++r) {
        int m = mt * 16 + r + 8 * lg;
        O[m * G3 + nt0 * 16 + ln]      = acc0[r];
        O[m * G3 + nt0 * 16 + 16 + ln] = acc1[r];
      }
    }
    gsync(p.flags, ++ep);

    // ---- phase 2: GRU0 elementwise (rank-1 prev term folded in) ----
    for (int i = gtid; i < BATCH * HD; i += NB * TPB) {
      int b = i >> 9, j = i & 511;
      float pv = p.prev[b];
      const float* gc = p.gi0c + b * G3;
      const float* gh = p.gh0  + b * G3;
      float ir = gc[j]        + pv * p.wc[j];
      float iz = gc[512 + j]  + pv * p.wc[512 + j];
      float in_= gc[1024 + j] + pv * p.wc[1024 + j];
      float hr = gh[j]        + p.bhh0[j];
      float hz = gh[512 + j]  + p.bhh0[512 + j];
      float hn = gh[1024 + j] + p.bhh0[1024 + j];
      float r = sigmoidf_(ir + hr);
      float z = sigmoidf_(iz + hz);
      float n = tanhf(in_ + r * hn);
      float hv = (1.0f - z) * n + z * p.h0[i];
      p.h0[i] = hv;
      p.h0f[fragIndex(b, j, KT)] = (_Float16)hv;
    }
    gsync(p.flags, ++ep);

    // ---- phase 3: gi1 = h0n @ Wih1^T (stage fresh h0n into h0 slot) ----
    if (tid < 32) {
      tdm_load_lds_1d(sh, p.h0f, 16384);
      __builtin_amdgcn_s_wait_tensorcnt(0);
    }
    __syncthreads();
    for (int u = gw; u < 4 * 48; u += NWAVE) {
      int mt = u / 48, nt0 = (u % 48) * 2;
      v8f acc0 = {}, acc1 = {};
#pragma unroll
      for (int kt = 0; kt < KT; ++kt) {
        v16h a  = *(const v16h*)(sh      + ((((mt * KT) + kt) * 32 + lane) << 4));
        v16h b0 = *(const v16h*)(p.Wih1f + (((((nt0    ) * KT) + kt) * 32 + lane) << 4));
        v16h b1 = *(const v16h*)(p.Wih1f + (((((nt0 + 1) * KT) + kt) * 32 + lane) << 4));
        if (kt + 2 < KT)
          __builtin_prefetch((const void*)(p.Wih1f + (((((nt0) * KT) + kt + 2) * 32 + lane) << 4)), 0, 1);
        acc0 = wmma_acc(a, b0, acc0);
        acc1 = wmma_acc(a, b1, acc1);
      }
      int lg = lane >> 4, ln = lane & 15;
#pragma unroll
      for (int r = 0; r < 8; ++r) {
        int m = mt * 16 + r + 8 * lg;
        p.gi1[m * G3 + nt0 * 16 + ln]      = acc0[r];
        p.gi1[m * G3 + nt0 * 16 + 16 + ln] = acc1[r];
      }
    }
    gsync(p.flags, ++ep);

    // ---- phase 4: GRU1 elementwise ----
    for (int i = gtid; i < BATCH * HD; i += NB * TPB) {
      int b = i >> 9, j = i & 511;
      const float* gi = p.gi1 + b * G3;
      const float* gh = p.gh1 + b * G3;
      float r = sigmoidf_(gi[j] + p.bih1[j] + gh[j] + p.bhh1[j]);
      float z = sigmoidf_(gi[512 + j] + p.bih1[512 + j] + gh[512 + j] + p.bhh1[512 + j]);
      float n = tanhf(gi[1024 + j] + p.bih1[1024 + j] + r * (gh[1024 + j] + p.bhh1[1024 + j]));
      float hv = (1.0f - z) * n + z * p.h1[i];
      p.h1[i] = hv;
      p.h1f[fragIndex(b, j, KT)] = (_Float16)hv;
    }
    gsync(p.flags, ++ep);

    // ---- phase 5: y1 = relu(h1n @ Wo1^T + b_o1) (stage h1n into h1 slot) ----
    if (tid < 32) {
      tdm_load_lds_1d(sh + 32768, p.h1f, 16384);
      __builtin_amdgcn_s_wait_tensorcnt(0);
    }
    __syncthreads();
    for (int u = gw; u < 4 * 16; u += NWAVE) {
      int mt = u / 16, nt0 = (u % 16) * 2;
      const _Float16* A = sh + 32768;
      v8f acc0 = {}, acc1 = {};
#pragma unroll
      for (int kt = 0; kt < KT; ++kt) {
        v16h a  = *(const v16h*)(A      + ((((mt * KT) + kt) * 32 + lane) << 4));
        v16h b0 = *(const v16h*)(p.Wo1f + (((((nt0    ) * KT) + kt) * 32 + lane) << 4));
        v16h b1 = *(const v16h*)(p.Wo1f + (((((nt0 + 1) * KT) + kt) * 32 + lane) << 4));
        acc0 = wmma_acc(a, b0, acc0);
        acc1 = wmma_acc(a, b1, acc1);
      }
      int lg = lane >> 4, ln = lane & 15;
#pragma unroll
      for (int r = 0; r < 8; ++r) {
        int m = mt * 16 + r + 8 * lg;
        int n0 = nt0 * 16 + ln;
        p.y1[m * HD + n0]      = fmaxf(acc0[r] + p.bo1[n0], 0.0f);
        p.y1[m * HD + n0 + 16] = fmaxf(acc1[r] + p.bo1[n0 + 16], 0.0f);
      }
    }
    gsync(p.flags, ++ep);

    // ---- phase 6: y2[b] = relu(y1[b,:] . Wo2 + b_o2); feed back as prev ----
    if (gw < BATCH) {
      int b = gw;
      float s = 0.0f;
      for (int j = lane; j < HD; j += 32) s += p.y1[b * HD + j] * p.Wo2[j];
#pragma unroll
      for (int off = 16; off > 0; off >>= 1) s += __shfl_down(s, off, 32);
      if (lane == 0) {
        float v = fmaxf(s + p.bo2[0], 0.0f);
        p.out[b * TSTEPS + t] = v;
        p.prev[b] = v;
      }
    }
    gsync(p.flags, ++ep);
  }

  // h_i = stack([h0, h1]) -> appended after outputs
  for (int i = gtid; i < BATCH * HD; i += NB * TPB) {
    p.out[BATCH * TSTEPS + i]              = p.h0[i];
    p.out[BATCH * TSTEPS + BATCH * HD + i] = p.h1[i];
  }
}

extern "C" void kernel_launch(void* const* d_in, const int* in_sizes, int n_in,
                              void* d_out, int out_size, void* d_ws, size_t ws_size,
                              hipStream_t stream) {
  (void)in_sizes; (void)n_in; (void)out_size; (void)ws_size;
  Params p;
  p.ctx  = (const float*)d_in[0];
  // d_in[1] = expected_output (only defines T) ; d_in[14] = force (unused)
  p.Wih0 = (const float*)d_in[2];
  p.Whh0 = (const float*)d_in[3];
  p.bih0 = (const float*)d_in[4];
  p.bhh0 = (const float*)d_in[5];
  p.Wih1 = (const float*)d_in[6];
  p.Whh1 = (const float*)d_in[7];
  p.bih1 = (const float*)d_in[8];
  p.bhh1 = (const float*)d_in[9];
  p.Wo1  = (const float*)d_in[10];
  p.bo1  = (const float*)d_in[11];
  p.Wo2  = (const float*)d_in[12];
  p.bo2  = (const float*)d_in[13];
  p.out  = (float*)d_out;

  char* base = (char*)d_ws;
  size_t off = 0;
  auto take = [&](size_t bytes) -> void* {
    void* q = base + off;
    off += (bytes + 255) & ~(size_t)255;
    return q;
  };
  // zero-initialized prefix (one memset per launch)
  p.flags = (unsigned*)take(NB * sizeof(unsigned));
  p.prev  = (float*)take(BATCH * sizeof(float));
  p.h0    = (float*)take((size_t)BATCH * HD * sizeof(float));
  p.h1    = (float*)take((size_t)BATCH * HD * sizeof(float));
  p.h0f   = (_Float16*)take((size_t)BATCH * HD * 2);
  p.h1f   = (_Float16*)take((size_t)BATCH * HD * 2);
  size_t zero_bytes = off;
  // scratch fully overwritten before use each step
  p.y1    = (float*)take((size_t)BATCH * HD * sizeof(float));
  p.gh0   = (float*)take((size_t)BATCH * G3 * sizeof(float));
  p.gh1   = (float*)take((size_t)BATCH * G3 * sizeof(float));
  p.gi1   = (float*)take((size_t)BATCH * G3 * sizeof(float));
  p.gi0c  = (float*)take((size_t)BATCH * G3 * sizeof(float));
  p.wc    = (float*)take((size_t)G3 * sizeof(float));
  p.ctxf  = (_Float16*)take((size_t)BATCH * HD * 2);
  p.Wih0f = (_Float16*)take((size_t)G3 * HD * 2);
  p.Whh0f = (_Float16*)take((size_t)G3 * HD * 2);
  p.Wih1f = (_Float16*)take((size_t)G3 * HD * 2);
  p.Whh1f = (_Float16*)take((size_t)G3 * HD * 2);
  p.Wo1f  = (_Float16*)take((size_t)HD * HD * 2);

  hipMemsetAsync(d_ws, 0, zero_bytes, stream);
  hipLaunchKernelGGL(k_swizzle, dim3(256), dim3(TPB), 0, stream, p);
  hipLaunchKernelGGL(k_gemm_ctx, dim3(48), dim3(TPB), 0, stream, p);
  hipLaunchKernelGGL(k_main, dim3(NB), dim3(TPB), 131072 /*dyn LDS*/, stream, p);
}
